// QuantMLP_8375186227286
// MI455X (gfx1250) — 8-bit matrix operand order repaired and checked by reading, compiled, not run on hardware
//
#include <hip/hip_runtime.h>
#include <stdint.h>

typedef __attribute__((ext_vector_type(8))) int v8i;

namespace {
constexpr int kB = 64, kS = 197, kD = 1024, kH = 4096;
constexpr int kM = kB * kS;
constexpr size_t OFF_W1Q = 256;
constexpr size_t OFF_W2Q = OFF_W1Q + (size_t)kH * kD;
constexpr size_t OFF_XQ  = OFF_W2Q + (size_t)kD * kH;
constexpr size_t OFF_Y   = OFF_XQ  + (size_t)kM * kD;
constexpr size_t OFF_Q   = OFF_Y   + (size_t)kM * kH * 4;
constexpr int LDSP = 80;
}

union V8U { v8i v; int4 q[2]; int2 d[4]; };

__device__ __forceinline__ float wave_reduce_max(float v) {
#pragma unroll
  for (int off = 16; off > 0; off >>= 1)
    v = fmaxf(v, __shfl_down(v, off, 32));
  return v;
}

__global__ void k_init(unsigned* s) {
  if (threadIdx.x < 3) s[threadIdx.x] = 0u;
}

__global__ void k_maxabs(const float* __restrict__ w, int n, unsigned* __restrict__ slot) {
  int i = blockIdx.x * blockDim.x + threadIdx.x;
  int stride = gridDim.x * blockDim.x;
  float m = 0.0f;
  for (; i < n; i += stride) m = fmaxf(m, fabsf(w[i]));
  m = wave_reduce_max(m);
  if ((threadIdx.x & 31) == 0) atomicMax(slot, __float_as_uint(m));
}

__global__ void k_quantw(const float* __restrict__ w, int n4,
                         const unsigned* __restrict__ slot, unsigned* __restrict__ out) {
  float sw = __uint_as_float(*slot) * (1.0f / 127.0f);
  int i = blockIdx.x * blockDim.x + threadIdx.x;
  int stride = gridDim.x * blockDim.x;
  for (; i < n4; i += stride) {
    float4 v = ((const float4*)w)[i];
    int a = (int)fminf(fmaxf(rintf(v.x / sw), -127.0f), 127.0f);
    int b = (int)fminf(fmaxf(rintf(v.y / sw), -127.0f), 127.0f);
    int c = (int)fminf(fmaxf(rintf(v.z / sw), -127.0f), 127.0f);
    int d = (int)fminf(fmaxf(rintf(v.w / sw), -127.0f), 127.0f);
    out[i] = (unsigned)(a & 255) | ((unsigned)(b & 255) << 8) |
             ((unsigned)(c & 255) << 16) | ((unsigned)(d & 255) << 24);
  }
}

__global__ void k_packx(const int* __restrict__ x, int n4, unsigned* __restrict__ out) {
  int i = blockIdx.x * blockDim.x + threadIdx.x;
  int stride = gridDim.x * blockDim.x;
  for (; i < n4; i += stride) {
    int4 v = ((const int4*)x)[i];
    out[i] = (unsigned)(v.x & 255) | ((unsigned)(v.y & 255) << 8) |
             ((unsigned)(v.z & 255) << 16) | ((unsigned)(v.w & 255) << 24);
  }
}

__device__ __forceinline__ void gemm_core(const char* __restrict__ A,
                                          const char* __restrict__ B,
                                          int M, int K, int mBase, int nBase,
                                          v8i acc[4][4], char* ldsA, char* ldsB) {
  const int tid = threadIdx.x;
  const int lane = tid & 31;
  const int wave = tid >> 5;
  const int waveM = wave & 1;
  const int waveN = wave >> 1;
  const int l15 = lane & 15;
  const int half = lane >> 4;

  const v8i vzero = {0, 0, 0, 0, 0, 0, 0, 0};
#pragma unroll
  for (int i = 0; i < 4; ++i)
#pragma unroll
    for (int j = 0; j < 4; ++j) acc[i][j] = vzero;

  const int kTiles = K >> 6;
  for (int kt = 0; kt < kTiles; ++kt) {
    const int kBase = kt << 6;
    __syncthreads();
#pragma unroll
    for (int i = 0; i < 4; ++i) {
      int idx = tid + i * 128;
      int row = idx >> 2;
      int ch  = (idx & 3) << 4;
      int rg  = mBase + row;
      rg = rg < M ? rg : (M - 1);
      *(int4*)(ldsA + row * LDSP + ch) = *(const int4*)(A + (size_t)rg * K + kBase + ch);
      *(int4*)(ldsB + row * LDSP + ch) = *(const int4*)(B + (size_t)(nBase + row) * K + kBase + ch);
    }
    __syncthreads();

    v8i bfrag[4];
#pragma unroll
    for (int nt = 0; nt < 4; ++nt) {
      const char* p = ldsB + (waveN * 64 + nt * 16 + l15) * LDSP + half * 16;
      V8U u;
      u.q[0] = *(const int4*)(p);
      u.q[1] = *(const int4*)(p + 32);
      bfrag[nt] = u.v;
    }

#pragma unroll
    for (int mt = 0; mt < 4; ++mt) {
      const char* p = ldsA + (waveM * 64 + mt * 16 + l15) * LDSP + half * 16;
      V8U ua;
      ua.q[0] = *(const int4*)(p);
      ua.q[1] = *(const int4*)(p + 32);
#pragma unroll
      for (int nt = 0; nt < 4; ++nt) {
        acc[mt][nt] = __builtin_amdgcn_wmma_i32_16x16x64_iu8(
            true, ua.v, true, bfrag[nt], acc[mt][nt], false, false);
      }
    }
  }
}

__global__ void __launch_bounds__(128)
k_gemm1(const char* __restrict__ Aq, const char* __restrict__ Bq,
        const float* __restrict__ b1, const float* __restrict__ sx,
        const unsigned* __restrict__ scal, float* __restrict__ Y,
        unsigned* __restrict__ maxy) {
  __shared__ char ldsA[128 * LDSP];
  __shared__ char ldsB[128 * LDSP];
  const int mBase = blockIdx.y * 128;
  const int nBase = blockIdx.x * 128;

  v8i acc[4][4];
  gemm_core(Aq, Bq, kM, kD, mBase, nBase, acc, ldsA, ldsB);

  const float sw1  = __uint_as_float(scal[0]) * (1.0f / 127.0f);
  const float s1   = sx[0] * sw1;
  const float sf   = s1 / 1.4142f;
  const float bint = floorf(-1.769f / sf);
  const float cint = floorf(-3.4626038f / (sf * sf));
  const float sigsf = ((sf * sf) * -0.2888f) * 16384.0f;
  const float shift = floorf(1.0f / sigsf);
  const float negb  = -bint;

  const int lane = threadIdx.x & 31;
  const int wave = threadIdx.x >> 5;
  const int waveM = wave & 1, waveN = wave >> 1;
  const int l15 = lane & 15, half = lane >> 4;

  float bi[4];
#pragma unroll
  for (int nt = 0; nt < 4; ++nt) {
    int n = nBase + waveN * 64 + nt * 16 + l15;
    bi[nt] = rintf(b1[n] / s1);
  }

  float wmax = 0.0f;
#pragma unroll
  for (int mt = 0; mt < 4; ++mt) {
#pragma unroll
    for (int nt = 0; nt < 4; ++nt) {
      int n = nBase + waveN * 64 + nt * 16 + l15;
#pragma unroll
      for (int r = 0; r < 8; ++r) {
        int m = mBase + waveM * 64 + mt * 16 + half * 8 + r;
        float a = (float)acc[mt][nt][r] + bi[nt];
        float sg = (a > 0.0f) ? 1.0f : ((a < 0.0f) ? -1.0f : 0.0f);
        float ab = fminf(fabsf(a), negb);
        float t  = ab + bint;
        float y1 = floorf((sg * (t * t + cint)) * (1.0f / 16384.0f));
        float yv = a * (y1 + shift);
        wmax = fmaxf(wmax, fabsf(yv));
        if (m < kM) Y[(size_t)m * kH + n] = yv;
      }
    }
  }
  wmax = wave_reduce_max(wmax);
  if (lane == 0) atomicMax(maxy, __float_as_uint(wmax));
}

__global__ void k_requant(const float* __restrict__ Y, const unsigned* __restrict__ scal,
                          const float* __restrict__ sx, unsigned* __restrict__ Q, int n4) {
  const float sw1   = __uint_as_float(scal[0]) * (1.0f / 127.0f);
  const float s1    = sx[0] * sw1;
  const float sf    = s1 / 1.4142f;
  const float sigsf = ((sf * sf) * -0.2888f) * 16384.0f;
  const float sfout = (s1 * sigsf) * 0.5f;
  const float maxy  = __uint_as_float(scal[2]);
  const float sa    = fabsf(sfout) * maxy / 127.0f;

  int i = blockIdx.x * blockDim.x + threadIdx.x;
  int stride = gridDim.x * blockDim.x;
  for (; i < n4; i += stride) {
    float4 v = ((const float4*)Y)[i];
    int a = (int)fminf(fmaxf(rintf((v.x * sfout) / sa), -128.0f), 127.0f);
    int b = (int)fminf(fmaxf(rintf((v.y * sfout) / sa), -128.0f), 127.0f);
    int c = (int)fminf(fmaxf(rintf((v.z * sfout) / sa), -128.0f), 127.0f);
    int d = (int)fminf(fmaxf(rintf((v.w * sfout) / sa), -128.0f), 127.0f);
    Q[i] = (unsigned)(a & 255) | ((unsigned)(b & 255) << 8) |
           ((unsigned)(c & 255) << 16) | ((unsigned)(d & 255) << 24);
  }
}

__global__ void __launch_bounds__(128)
k_gemm2(const char* __restrict__ Aq, const char* __restrict__ Bq,
        const float* __restrict__ b2, const float* __restrict__ sx,
        const unsigned* __restrict__ scal, float* __restrict__ out) {
  __shared__ char ldsA[128 * LDSP];
  __shared__ char ldsB[128 * LDSP];
  const int mBase = blockIdx.y * 128;
  const int nBase = blockIdx.x * 128;

  v8i acc[4][4];
  gemm_core(Aq, Bq, kM, kH, mBase, nBase, acc, ldsA, ldsB);

  const float sw1   = __uint_as_float(scal[0]) * (1.0f / 127.0f);
  const float s1    = sx[0] * sw1;
  const float sf    = s1 / 1.4142f;
  const float sigsf = ((sf * sf) * -0.2888f) * 16384.0f;
  const float sfout = (s1 * sigsf) * 0.5f;
  const float maxy  = __uint_as_float(scal[2]);
  const float sa    = fabsf(sfout) * maxy / 127.0f;
  const float sw2   = __uint_as_float(scal[1]) * (1.0f / 127.0f);
  const float sout  = sa * sw2;

  const int lane = threadIdx.x & 31;
  const int wave = threadIdx.x >> 5;
  const int waveM = wave & 1, waveN = wave >> 1;
  const int l15 = lane & 15, half = lane >> 4;

  float bi[4];
#pragma unroll
  for (int nt = 0; nt < 4; ++nt) {
    int n = nBase + waveN * 64 + nt * 16 + l15;
    bi[nt] = rintf(b2[n] / sout);
  }

#pragma unroll
  for (int mt = 0; mt < 4; ++mt) {
#pragma unroll
    for (int nt = 0; nt < 4; ++nt) {
      int n = nBase + waveN * 64 + nt * 16 + l15;
#pragma unroll
      for (int r = 0; r < 8; ++r) {
        int m = mBase + waveM * 64 + mt * 16 + half * 8 + r;
        if (m < kM) out[(size_t)m * kD + n] = (float)acc[mt][nt][r] + bi[nt];
      }
    }
  }
  if (blockIdx.x == 0 && blockIdx.y == 0 && threadIdx.x == 0)
    out[(size_t)kM * kD] = sout;
}

extern "C" void kernel_launch(void* const* d_in, const int* in_sizes, int n_in,
                              void* d_out, int out_size, void* d_ws, size_t ws_size,
                              hipStream_t stream) {
  const int*   x  = (const int*)d_in[0];
  const float* sx = (const float*)d_in[1];
  const float* w1 = (const float*)d_in[2];
  const float* b1 = (const float*)d_in[3];
  const float* w2 = (const float*)d_in[4];
  const float* b2 = (const float*)d_in[5];
  float* out = (float*)d_out;

  char* ws = (char*)d_ws;
  unsigned* scal = (unsigned*)ws;
  char*  w1q = ws + OFF_W1Q;
  char*  w2q = ws + OFF_W2Q;
  char*  xq  = ws + OFF_XQ;
  float* Y   = (float*)(ws + OFF_Y);
  char*  Q   = ws + OFF_Q;

  k_init<<<1, 32, 0, stream>>>(scal);
  k_maxabs<<<1024, 256, 0, stream>>>(w1, kH * kD, scal + 0);
  k_maxabs<<<1024, 256, 0, stream>>>(w2, kD * kH, scal + 1);
  k_quantw<<<2048, 256, 0, stream>>>(w1, (kH * kD) / 4, scal + 0, (unsigned*)w1q);
  k_quantw<<<2048, 256, 0, stream>>>(w2, (kD * kH) / 4, scal + 1, (unsigned*)w2q);
  k_packx<<<2048, 256, 0, stream>>>(x, (kM * kD) / 4, (unsigned*)xq);

  dim3 g1(kH / 128, (kM + 127) / 128);
  k_gemm1<<<g1, 128, 0, stream>>>(xq, w1q, b1, sx, scal, Y, scal + 2);

  k_requant<<<4096, 256, 0, stream>>>(Y, scal, sx, (unsigned*)Q, (kM * kH) / 4);

  dim3 g2(kD / 128, (kM + 127) / 128);
  k_gemm2<<<g2, 128, 0, stream>>>(Q, w2q, b2, sx, scal, out);
}
